// LSTMLayer_24163486007734
// MI455X (gfx1250) — matrix operand order repaired and checked by reading, not compiled, not run on hardware
//
#include <hip/hip_runtime.h>

// ---------------------------------------------------------------- constants
static constexpr int T  = 256;
static constexpr int B  = 64;
static constexpr int I  = 512;
static constexpr int H  = 1024;
static constexpr int G  = 4 * H;        // 4096 gate columns
static constexpr int NT = G / 16;       // 256 n-tiles per row-block
static constexpr int TILE_ELEMS = 256;  // 16x16

typedef unsigned short u16;
typedef unsigned int   u32;
typedef __attribute__((ext_vector_type(16))) __bf16 v16bf;
typedef __attribute__((ext_vector_type(8)))  float  v8f;
typedef __attribute__((ext_vector_type(4)))  float  f32x4;
typedef __attribute__((ext_vector_type(4)))  u32    u32x4;
typedef __attribute__((ext_vector_type(4)))  int    i32x4;

#if defined(__has_builtin)
#if __has_builtin(__builtin_amdgcn_global_load_async_to_lds_b128) && \
    __has_builtin(__builtin_amdgcn_s_wait_asynccnt)
#define HAVE_ASYNC_LDS 1
#endif
#endif
#ifndef HAVE_ASYNC_LDS
#define HAVE_ASYNC_LDS 0
#endif

#if HAVE_ASYNC_LDS
typedef __attribute__((address_space(1))) i32x4 g_i32x4;   // global int4
typedef __attribute__((address_space(3))) i32x4 l_i32x4;   // LDS int4
#endif

// ---------------------------------------------------------------- helpers
__device__ __forceinline__ u16 f32_to_bf16(float f) {
  union { float f; u32 u; } x; x.f = f;
  u32 r = x.u + 0x7FFFu + ((x.u >> 16) & 1u);   // round-to-nearest-even
  return (u16)(r >> 16);
}
__device__ __forceinline__ float bf16_to_f32(u16 h) {
  union { u32 u; float f; } x; x.u = ((u32)h) << 16;
  return x.f;
}
__device__ __forceinline__ float fast_sigmoid(float x) {
  x = fminf(fmaxf(x, -30.f), 30.f);
  return 1.0f / (1.0f + __expf(-x));
}
__device__ __forceinline__ float fast_tanh(float x) {
  x = fminf(fmaxf(x, -15.f), 15.f);
  float e = __expf(2.0f * x);
  return (e - 1.0f) / (e + 1.0f);
}

// A fragment: 16x32 (MxK) bf16, row-major source, base = &src[mbase*ld + kbase].
// Works for global or LDS pointers (addrspace inferred after inlining).
__device__ __forceinline__ v16bf load_a_frag(const u16* base, int lane, int ld) {
  int half = lane >> 4;
  int m    = lane & 15;
  const u16* p = base + (size_t)m * ld + half * 8;
  union { v16bf v; u32x4 q[2]; } f;
  f.q[0] = *(const u32x4*)(p);        // K = half*8 .. +7      -> v0..v3
  f.q[1] = *(const u32x4*)(p + 16);   // K = 16+half*8 .. +7   -> v4..v7
  return f.v;
}

// B fragment: 32x16 (KxN) bf16 from N-major storage W[n][k], base = &W[nbase*ld + kbase].
__device__ __forceinline__ v16bf load_b_frag(const u16* __restrict__ base, int lane, int ld) {
  int n  = lane & 15;
  int kh = (lane >> 4) * 8;
  const u16* p = base + (size_t)n * ld + kh;
  union { v16bf v; u32x4 q[2]; } f;
  f.q[0] = *(const u32x4*)(p);
  f.q[1] = *(const u32x4*)(p + 16);
  return f.v;
}

template <bool RA, bool RB>
__device__ __forceinline__ v8f wmma_bf16(v16bf a, v16bf b, v8f c) {
  return __builtin_amdgcn_wmma_f32_16x16x32_bf16(false, a, false, b, (short)0, c, RA, RB);
}

// fragment-major f32 gather index for a [64 x G] matrix stored as
// [mi(4)][nt(256)][lane(32)][r(8)]
__device__ __forceinline__ size_t gf_index(int b, int ng) {
  int mi   = b >> 4;
  int nt   = ng >> 4;
  int lane = (ng & 15) | (((b >> 3) & 1) << 4);
  int r    = b & 7;
  return ((size_t)(mi * NT + nt)) * TILE_ELEMS + lane * 8 + r;
}

// ---------------------------------------------------------------- prep kernels
__global__ void prep_wh_kernel(const float* __restrict__ wh, const float* __restrict__ mask,
                               u16* __restrict__ whb, int n) {
  for (int i = blockIdx.x * blockDim.x + threadIdx.x; i < n; i += gridDim.x * blockDim.x)
    whb[i] = f32_to_bf16(wh[i] * mask[i]);
}
__global__ void cvt_bf16_kernel(const float* __restrict__ src, u16* __restrict__ dst, int n) {
  for (int i = blockIdx.x * blockDim.x + threadIdx.x; i < n; i += gridDim.x * blockDim.x)
    dst[i] = f32_to_bf16(src[i]);
}
__global__ void bias_sum_kernel(const float* __restrict__ b1, const float* __restrict__ b2,
                                float* __restrict__ bs, int n) {
  for (int i = blockIdx.x * blockDim.x + threadIdx.x; i < n; i += gridDim.x * blockDim.x)
    bs[i] = b1[i] + b2[i];
}
__global__ void init_state_kernel(const float* __restrict__ h0, const float* __restrict__ c0,
                                  u16* __restrict__ hb, float* __restrict__ cst, int n) {
  for (int i = blockIdx.x * blockDim.x + threadIdx.x; i < n; i += gridDim.x * blockDim.x) {
    hb[i]  = f32_to_bf16(h0[i]);
    cst[i] = c0[i];
  }
}

// ---------------------------------------------------------------- x-projection GEMM
// XPf (bf16, fragment-major [mt(1024)][nt(256)][lane][8]) =
//      Xb[T*B][I] @ Wib^T + bias_sum   (Wib stored N-major [G][I])
__global__ void xproj_gemm_kernel(const u16* __restrict__ Xb, const u16* __restrict__ Wib,
                                  const float* __restrict__ bias, u16* __restrict__ XPf) {
  int gwave = (blockIdx.x * blockDim.x + threadIdx.x) >> 5;
  int lane  = threadIdx.x & 31;
  const int NG = G / 64;                 // 64 n-groups of 4 tiles
  int mt = gwave / NG;                   // 0..1023
  int ng = gwave % NG;
  int mbase = mt * 16;
  int nbase = ng * 64;

  const u16* Ab  = Xb  + (size_t)mbase * I;
  const u16* Bb0 = Wib + (size_t)(nbase +  0) * I;
  const u16* Bb1 = Wib + (size_t)(nbase + 16) * I;
  const u16* Bb2 = Wib + (size_t)(nbase + 32) * I;
  const u16* Bb3 = Wib + (size_t)(nbase + 48) * I;

  v8f c0 = {}, c1 = {}, c2 = {}, c3 = {};

  v16bf a_c = load_a_frag(Ab,  lane, I);
  v16bf bc0 = load_b_frag(Bb0, lane, I);
  v16bf bc1 = load_b_frag(Bb1, lane, I);
  v16bf bc2 = load_b_frag(Bb2, lane, I);
  v16bf bc3 = load_b_frag(Bb3, lane, I);

  constexpr int KIT = I / 32;            // 16
  #pragma unroll
  for (int kk = 0; kk < KIT - 1; ++kk) {
    int ko = (kk + 1) * 32;
    v16bf a_n = load_a_frag(Ab  + ko, lane, I);
    v16bf bn0 = load_b_frag(Bb0 + ko, lane, I);
    v16bf bn1 = load_b_frag(Bb1 + ko, lane, I);
    v16bf bn2 = load_b_frag(Bb2 + ko, lane, I);
    v16bf bn3 = load_b_frag(Bb3 + ko, lane, I);
    c0 = wmma_bf16<false, false>(a_c, bc0, c0);
    c1 = wmma_bf16<true,  false>(a_c, bc1, c1);   // reuse A
    c2 = wmma_bf16<true,  false>(a_c, bc2, c2);
    c3 = wmma_bf16<true,  false>(a_c, bc3, c3);
    a_c = a_n; bc0 = bn0; bc1 = bn1; bc2 = bn2; bc3 = bn3;
  }
  c0 = wmma_bf16<false, false>(a_c, bc0, c0);
  c1 = wmma_bf16<true,  false>(a_c, bc1, c1);
  c2 = wmma_bf16<true,  false>(a_c, bc2, c2);
  c3 = wmma_bf16<true,  false>(a_c, bc3, c3);

  int n = lane & 15;
  v8f acc[4] = { c0, c1, c2, c3 };
  #pragma unroll
  for (int j = 0; j < 4; ++j) {
    float bv = bias[nbase + j * 16 + n];
    union { u32x4 q; u16 s[8]; } pk;
    #pragma unroll
    for (int r = 0; r < 8; ++r) pk.s[r] = f32_to_bf16(acc[j][r] + bv);
    size_t toff = ((size_t)mt * NT + ng * 4 + j) * TILE_ELEMS + lane * 8;
    *(u32x4*)(XPf + toff) = pk.q;
  }
}

// ---------------------------------------------------------------- recurrent gate GEMM (one step)
// Gf[4][256 tiles] (frag-major f32) = XPf(t) + Hb[B][H] @ Whb^T
// h-state (128 KB bf16) staged into LDS (async DMA when available); A fragments
// from LDS (DScnt), B weight fragments from L2 with prefetch distance 2 (LOADcnt).
// One wave -> 64-row x 16-col strip (4 tiles). 256 waves, 32 blocks.
__global__ void lstm_gates_gemm_kernel(const u16* __restrict__ Hb, const u16* __restrict__ Whb,
                                       const u16* __restrict__ XPt, float* __restrict__ Gf) {
  extern __shared__ u16 sh[];            // [B][H] bf16 = 128 KB

  // -------- stage h-state into LDS
  {
    const int nchunks = (B * H) / 8;     // 8192 x 16B
#if HAVE_ASYNC_LDS
    g_i32x4* gsrc = (g_i32x4*)(u16*)Hb;
    l_i32x4* ldst = (l_i32x4*)sh;
    for (int i = threadIdx.x; i < nchunks; i += blockDim.x)
      __builtin_amdgcn_global_load_async_to_lds_b128(gsrc + i, ldst + i, 0, 0);
    __builtin_amdgcn_s_wait_asynccnt(0);
#else
    const u32x4* src = (const u32x4*)Hb;
    u32x4*       dst = (u32x4*)sh;
    for (int i = threadIdx.x; i < nchunks; i += blockDim.x)
      dst[i] = src[i];
#endif
  }
  __syncthreads();

  int wv   = blockIdx.x * (blockDim.x >> 5) + (threadIdx.x >> 5);  // 0..255
  int lane = threadIdx.x & 31;
  int nbase = wv * 16;

  const u16* Bb = Whb + (size_t)nbase * H;

  // init C from fragment-major bf16 x-projection (one b128 per tile)
  v8f c0, c1, c2, c3;
  {
    union { u32x4 q; u16 s[8]; } p0, p1, p2, p3;
    p0.q = *(const u32x4*)(XPt + ((size_t)(0 * NT + wv)) * TILE_ELEMS + lane * 8);
    p1.q = *(const u32x4*)(XPt + ((size_t)(1 * NT + wv)) * TILE_ELEMS + lane * 8);
    p2.q = *(const u32x4*)(XPt + ((size_t)(2 * NT + wv)) * TILE_ELEMS + lane * 8);
    p3.q = *(const u32x4*)(XPt + ((size_t)(3 * NT + wv)) * TILE_ELEMS + lane * 8);
    #pragma unroll
    for (int r = 0; r < 8; ++r) {
      c0[r] = bf16_to_f32(p0.s[r]); c1[r] = bf16_to_f32(p1.s[r]);
      c2[r] = bf16_to_f32(p2.s[r]); c3[r] = bf16_to_f32(p3.s[r]);
    }
  }

  // -------- k-loop: B stream prefetch distance 2, A from LDS
  constexpr int KIT = H / 32;            // 32
  v16bf b0 = load_b_frag(Bb,      lane, H);
  v16bf b1 = load_b_frag(Bb + 32, lane, H);

  #pragma unroll
  for (int kk = 0; kk < KIT - 2; ++kk) {
    v16bf b2 = load_b_frag(Bb + (kk + 2) * 32, lane, H);
    v16bf a0 = load_a_frag(sh + (size_t)( 0) * H + kk * 32, lane, H);
    v16bf a1 = load_a_frag(sh + (size_t)(16) * H + kk * 32, lane, H);
    v16bf a2 = load_a_frag(sh + (size_t)(32) * H + kk * 32, lane, H);
    v16bf a3 = load_a_frag(sh + (size_t)(48) * H + kk * 32, lane, H);
    c0 = wmma_bf16<false, false>(a0, b0, c0);
    c1 = wmma_bf16<false, true >(a1, b0, c1);   // reuse B
    c2 = wmma_bf16<false, true >(a2, b0, c2);
    c3 = wmma_bf16<false, true >(a3, b0, c3);
    b0 = b1; b1 = b2;
  }
  #pragma unroll
  for (int kk = KIT - 2; kk < KIT; ++kk) {
    v16bf a0 = load_a_frag(sh + (size_t)( 0) * H + kk * 32, lane, H);
    v16bf a1 = load_a_frag(sh + (size_t)(16) * H + kk * 32, lane, H);
    v16bf a2 = load_a_frag(sh + (size_t)(32) * H + kk * 32, lane, H);
    v16bf a3 = load_a_frag(sh + (size_t)(48) * H + kk * 32, lane, H);
    c0 = wmma_bf16<false, false>(a0, b0, c0);
    c1 = wmma_bf16<false, true >(a1, b0, c1);
    c2 = wmma_bf16<false, true >(a2, b0, c2);
    c3 = wmma_bf16<false, true >(a3, b0, c3);
    b0 = b1;
  }

  // fragment-major f32 store: two f32x4 per tile
  v8f acc[4] = { c0, c1, c2, c3 };
  #pragma unroll
  for (int mi = 0; mi < 4; ++mi) {
    size_t o = ((size_t)(mi * NT + wv)) * TILE_ELEMS + lane * 8;
    f32x4 lo = { acc[mi][0], acc[mi][1], acc[mi][2], acc[mi][3] };
    f32x4 hi = { acc[mi][4], acc[mi][5], acc[mi][6], acc[mi][7] };
    *(f32x4*)(Gf + o)     = lo;
    *(f32x4*)(Gf + o + 4) = hi;
  }
}

// ---------------------------------------------------------------- pointwise LSTM cell (one step)
__global__ void lstm_pointwise_kernel(const float* __restrict__ Gf, float* __restrict__ cst,
                                      u16* __restrict__ hb, float* __restrict__ out_h,
                                      float* __restrict__ hT, float* __restrict__ cT,
                                      int writeFinal) {
  int idx = blockIdx.x * blockDim.x + threadIdx.x;   // 0 .. B*H-1
  int b = idx >> 10;                                  // H = 1024
  int h = idx & (H - 1);
  float gi = Gf[gf_index(b,         h)];
  float gf = Gf[gf_index(b,     H + h)];
  float gg = Gf[gf_index(b, 2 * H + h)];
  float go = Gf[gf_index(b, 3 * H + h)];
  float cn = fast_sigmoid(gf) * cst[idx] + fast_sigmoid(gi) * fast_tanh(gg);
  float hn = fast_sigmoid(go) * fast_tanh(cn);
  cst[idx]   = cn;
  out_h[idx] = hn;
  hb[idx]    = f32_to_bf16(hn);
  if (writeFinal) { hT[idx] = hn; cT[idx] = cn; }
}

// ---------------------------------------------------------------- launcher
extern "C" void kernel_launch(void* const* d_in, const int* in_sizes, int n_in,
                              void* d_out, int out_size, void* d_ws, size_t ws_size,
                              hipStream_t stream) {
  const float* input     = (const float*)d_in[0];   // [T,B,I]
  const float* h0        = (const float*)d_in[1];   // [B,H]
  const float* c0        = (const float*)d_in[2];   // [B,H]
  const float* w_mask    = (const float*)d_in[3];   // [4H,H]
  const float* weight_ih = (const float*)d_in[4];   // [4H,I]
  const float* weight_hh = (const float*)d_in[5];   // [4H,H]
  const float* bias_ih   = (const float*)d_in[6];   // [4H]
  const float* bias_hh   = (const float*)d_in[7];   // [4H]

  float* outputs = (float*)d_out;                   // [T,B,H]
  float* hT      = outputs + (size_t)T * B * H;     // [B,H]
  float* cT      = hT + (size_t)B * H;              // [B,H]

  // workspace carve-up (256B aligned)
  char* ws = (char*)d_ws;
  size_t off = 0;
  auto take = [&](size_t bytes) { char* p = ws + off; off = (off + bytes + 255) & ~(size_t)255; return p; };
  u16*   Whb  = (u16*)  take((size_t)G * H * 2);      // masked hh weights, bf16, N-major [G][H]
  u16*   Wib  = (u16*)  take((size_t)G * I * 2);      // ih weights, bf16, N-major [G][I]
  float* bsum = (float*)take((size_t)G * 4);          // bias_ih + bias_hh
  u16*   Xb   = (u16*)  take((size_t)T * B * I * 2);  // input, bf16, row-major [T*B][I]
  u16*   XPf  = (u16*)  take((size_t)T * B * G * 2);  // x-projection, bf16, fragment-major
  u16*   Hb   = (u16*)  take((size_t)B * H * 2);      // h state, bf16 [B][H]
  float* Cst  = (float*)take((size_t)B * H * 4);      // c state, f32
  float* Gts  = (float*)take((size_t)B * G * 4);      // per-step gates, f32, fragment-major
  (void)ws_size; (void)n_in; (void)in_sizes; (void)out_size;

  // ---- prep
  prep_wh_kernel   <<<1024, 256, 0, stream>>>(weight_hh, w_mask, Whb, G * H);
  cvt_bf16_kernel  <<<1024, 256, 0, stream>>>(weight_ih, Wib, G * I);
  bias_sum_kernel  <<<16,   256, 0, stream>>>(bias_ih, bias_hh, bsum, G);
  cvt_bf16_kernel  <<<1024, 256, 0, stream>>>(input, Xb, T * B * I);
  init_state_kernel<<<256,  256, 0, stream>>>(h0, c0, Hb, Cst, B * H);

  // ---- input projection: 1024 m-tiles x 64 n-groups = 65536 waves, 8 waves/block
  xproj_gemm_kernel<<<8192, 256, 0, stream>>>(Xb, Wib, bsum, XPf);

  // ---- sequential scan: 256 waves (64x16 strip each), h-state staged in LDS
  const size_t lds_bytes = (size_t)B * H * sizeof(u16);   // 128 KB dynamic LDS
  for (int t = 0; t < T; ++t) {
    const u16* XPt = XPf + (size_t)t * 4 * NT * TILE_ELEMS;   // t-th [64 x G] block
    lstm_gates_gemm_kernel<<<32, 256, lds_bytes, stream>>>(Hb, Whb, XPt, Gts);
    lstm_pointwise_kernel <<<256, 256, 0, stream>>>(Gts, Cst, Hb,
                                                    outputs + (size_t)t * B * H,
                                                    hT, cT, (t == T - 1) ? 1 : 0);
  }
}
